// KernelSample_72971494359165
// MI455X (gfx1250) — matrix operand order repaired and checked by reading, not compiled, not run on hardware
//
#include <hip/hip_runtime.h>

typedef _Float16 half_t;
typedef __attribute__((ext_vector_type(16))) _Float16 v16h;
typedef __attribute__((ext_vector_type(8)))  float    v8f;
typedef __attribute__((ext_vector_type(4)))  unsigned int u32x4;
typedef __attribute__((ext_vector_type(8)))  int          i32x8;
typedef __attribute__((ext_vector_type(4)))  int          i32x4;

#define N_PTS   4096
#define DIMW    64
#define BC      8
#define KC      64                         // input points per chunk
#define NCHUNK  (N_PTS / KC)
#define M_TILE  128                        // rows per block = 8 waves x 16

#define WCOL_HALFS 72                      // 64 data + 8 pad halfs per column (144B, conflict-free)
#define WBUF_BYTES (DIMW * WCOL_HALFS * 2) // 9216 per buffer
#define XSBUF_BYTES (KC * 16)              // 1024 per buffer: 4 SoA rows x 64 f32
#define LDS_XS_OFF (2 * WBUF_BYTES)

// exp(-d/2) = exp2(-d * log2(e)/2)
#define NEG_HALF_LOG2E (-0.7213475204444817f)

// d_ws layout
#define WS_WT_BYTES ((size_t)BC * DIMW * N_PTS * 2)  // f16 weights, (bc, col, k)
#define WS_XS_OFF   WS_WT_BYTES                      // input pos SoA: (bc, 4, 4096) f32
#define WS_XS_BYTES ((size_t)BC * 4 * N_PTS * 4)
#define WS_YS_OFF   (WS_XS_OFF + WS_XS_BYTES)        // output pos AoS float4 (x,y,z,sq)

// ---------------- prep kernels ----------------

__global__ __launch_bounds__(256) void prep_weights(const float* __restrict__ w,
                                                    half_t* __restrict__ wT) {
    // wT[bc][c][k] = (f16) w[bc][k][c]
    const size_t idx = (size_t)blockIdx.x * 256 + threadIdx.x;   // BC*DIMW*N_PTS
    const int k  = (int)(idx & (N_PTS - 1));
    const int c  = (int)((idx >> 12) & (DIMW - 1));
    const int bc = (int)(idx >> 18);
    wT[idx] = (half_t)w[((size_t)bc * N_PTS + k) * DIMW + c];
}

__global__ __launch_bounds__(256) void prep_pos_soa(const float* __restrict__ p,
                                                    float* __restrict__ o) {
    // o[bc][comp][k], comp = x,y,z,|p|^2
    const int i  = blockIdx.x * 256 + threadIdx.x;               // BC*N_PTS
    const int bc = i >> 12;
    const int k  = i & (N_PTS - 1);
    const float a = p[3 * i + 0], b = p[3 * i + 1], c = p[3 * i + 2];
    float* base = o + (size_t)bc * 4 * N_PTS;
    base[0 * N_PTS + k] = a;
    base[1 * N_PTS + k] = b;
    base[2 * N_PTS + k] = c;
    base[3 * N_PTS + k] = a * a + b * b + c * c;
}

__global__ __launch_bounds__(256) void prep_pos_aos(const float* __restrict__ p,
                                                    float4* __restrict__ o) {
    const int i = blockIdx.x * 256 + threadIdx.x;                // BC*N_PTS
    const float a = p[3 * i + 0], b = p[3 * i + 1], c = p[3 * i + 2];
    o[i] = make_float4(a, b, c, a * a + b * b + c * c);
}

// ---------------- TDM descriptor issue (ISA 8.3/8.4) ----------------

__device__ __forceinline__ void tdm_issue(unsigned lds_off, unsigned long long gaddr,
                                          int g1w0, int g1w1, int g1w2, int g1w3,
                                          int g1w4, int g1w5) {
    u32x4 g0;
    g0[0] = 1u;                                              // count=1, user mode
    g0[1] = lds_off;                                         // LDS byte address
    g0[2] = (unsigned)(gaddr & 0xFFFFFFFFu);                 // global_addr[31:0]
    g0[3] = (unsigned)((gaddr >> 32) & 0x1FFFFFFu) | (2u << 30); // [56:32] | type=2
    i32x8 g1 = { g1w0, g1w1, g1w2, g1w3, g1w4, g1w5, 0, 0 };
    i32x4 z4 = { 0, 0, 0, 0 };
    i32x8 z8 = { 0, 0, 0, 0, 0, 0, 0, 0 };
    __builtin_amdgcn_tensor_load_to_lds(g0, g1, z4, z4, z8, 0);
}

// ---------------- main kernel ----------------

__global__ __launch_bounds__(256) void gauss_gram_wmma(
    const half_t* __restrict__ wT,    // (BC, 64, 4096) f16
    const float*  __restrict__ xsoa,  // (BC, 4, 4096) x|y|z|sq
    const float4* __restrict__ ys4,   // (BC, 4096) x,y,z,sq
    float* __restrict__ out)          // (BC, 4096, 64)
{
    __shared__ __align__(16) unsigned char smem[2 * WBUF_BYTES + 2 * XSBUF_BYTES];

    const int bc   = blockIdx.y;
    const int tid  = threadIdx.x;
    const int lane = tid & 31;
    const int wave = tid >> 5;
    const int o_base = blockIdx.x * M_TILE + wave * 16;

    // Per-lane output point (row m of this wave's 16-row stripe).
    const int m = lane & 15;
    const float4 yv = ys4[bc * N_PTS + o_base + m];
    const float y0 = yv.x, y1 = yv.y, y2 = yv.z, sqy = yv.w;

    // A operand, this code's k order: half h -> K = kbase_a + h (contiguous)
    const int kbase_a = (lane < 16) ? 0 : 16;
    // B operand, same k order as A: half h -> K = kb_half + h (contiguous)
    const int kb_half = (lane < 16) ? 0 : 16;

    const unsigned long long wg = (unsigned long long)(uintptr_t)(wT + (size_t)bc * DIMW * N_PTS);
    const unsigned long long xg = (unsigned long long)(uintptr_t)(xsoa + (size_t)bc * 4 * N_PTS);
    const unsigned lds_base = (unsigned)(uintptr_t)&smem[0];   // low 32 bits = LDS offset

    v8f accs[4] = {{}, {}, {}, {}};

    // Issue one TDM pair (weights tile + positions tile) for chunk t into buffer b.
    auto issue = [&](int t, int b) {
        // Weights: 2D tensor (dim0=k len 4096 stride 4096, dim1=col len 64),
        // tile 64x64 halfs, LDS pad 4 DWORDs after every 32 DWORDs (row) -> 144B col stride.
        tdm_issue(lds_base + b * WBUF_BYTES,
                  wg + (unsigned long long)t * KC * 2,
                  (1 << 16) | (1 << 20) | (4 << 22) | (3 << 25), // data_size=2B, pad_en, int=32dw, amt=4dw
                  (int)((N_PTS & 0xFFFF) << 16),                 // tensor_dim0 lo
                  (int)(DIMW << 16),                             // dim0 hi=0 | tensor_dim1
                  (int)(KC << 16),                               // dim1 hi=0 | tile_dim0
                  DIMW,                                          // tile_dim1 | tile_dim2=0
                  N_PTS);                                        // tensor_dim0_stride lo
        // Positions SoA: 2D tensor (dim0=k len 4096 stride 4096, dim1=4 comps),
        // tile 64x4 f32 -> LDS as 4 contiguous rows of 256B.
        tdm_issue(lds_base + LDS_XS_OFF + b * XSBUF_BYTES,
                  xg + (unsigned long long)t * KC * 4,
                  (2 << 16),                                     // data_size=4B
                  (int)((N_PTS & 0xFFFF) << 16),                 // tensor_dim0 lo
                  (int)(4 << 16),                                // dim0 hi=0 | tensor_dim1=4
                  (int)(KC << 16),                               // dim1 hi=0 | tile_dim0=64
                  4,                                             // tile_dim1=4 | tile_dim2=0
                  N_PTS);                                        // tensor_dim0_stride lo
    };

    if (wave == 0) issue(0, 0);

    for (int t = 0; t < NCHUNK; ++t) {
        const int buf = t & 1;
        if (wave == 0) {
            if (t + 1 < NCHUNK) {
                issue(t + 1, buf ^ 1);
                __builtin_amdgcn_s_wait_tensorcnt((short)2);   // chunk t's 2 ops done (in-order)
            } else {
                __builtin_amdgcn_s_wait_tensorcnt((short)0);
            }
        }
        __syncthreads();   // chunk t visible to all waves

        const float* xx = (const float*)(smem + LDS_XS_OFF + buf * XSBUF_BYTES);
        const float* xy = xx + KC;
        const float* xz = xx + 2 * KC;
        const float* xq = xx + 3 * KC;
        const half_t* wb = (const half_t*)(smem + buf * WBUF_BYTES);

        #pragma unroll
        for (int s = 0; s < 2; ++s) {
            // Build A: 16x32 Gaussian tile directly in WMMA A layout.
            // Quad q covers halfs h = 4q..4q+3 -> K = kk..kk+3 (contiguous).
            v16h a;
            #pragma unroll
            for (int q = 0; q < 4; ++q) {
                const int kk = s * 32 + kbase_a + q * 4;
                const float4 X = *(const float4*)(xx + kk);    // 4 points' x  (ds_read_b128)
                const float4 Y = *(const float4*)(xy + kk);
                const float4 Z = *(const float4*)(xz + kk);
                const float4 Q = *(const float4*)(xq + kk);
                const float d0 = sqy + Q.x - 2.0f * (y0 * X.x + y1 * Y.x + y2 * Z.x);
                const float d1 = sqy + Q.y - 2.0f * (y0 * X.y + y1 * Y.y + y2 * Z.y);
                const float d2 = sqy + Q.z - 2.0f * (y0 * X.z + y1 * Y.z + y2 * Z.z);
                const float d3 = sqy + Q.w - 2.0f * (y0 * X.w + y1 * Y.w + y2 * Z.w);
                a[q * 4 + 0] = (half_t)__builtin_amdgcn_exp2f(NEG_HALF_LOG2E * d0);
                a[q * 4 + 1] = (half_t)__builtin_amdgcn_exp2f(NEG_HALF_LOG2E * d1);
                a[q * 4 + 2] = (half_t)__builtin_amdgcn_exp2f(NEG_HALF_LOG2E * d2);
                a[q * 4 + 3] = (half_t)__builtin_amdgcn_exp2f(NEG_HALF_LOG2E * d3);
            }
            // One A feeds all 4 column tiles (N = 64).
            #pragma unroll
            for (int ct = 0; ct < 4; ++ct) {
                const half_t* bp = wb + (ct * 16 + (lane & 15)) * WCOL_HALFS + s * 32 + kb_half;
                v16h b;                                        // 32B contiguous, 16B aligned
                __builtin_memcpy(&b, bp, sizeof(b));           // -> 2x ds_load_b128
                accs[ct] = __builtin_amdgcn_wmma_f32_16x16x32_f16(
                    false, a, false, b, (short)0, accs[ct], false, false);
            }
        }
        __syncthreads();   // everyone done with buf before TDM overwrites it next iter
    }

    // Writeback: C/D layout: VGPR r -> row M = r (+8 for lanes 16-31), col N = lane&15.
    float* op = out + ((size_t)bc * N_PTS + o_base) * DIMW;
    const int row_off = (lane < 16) ? 0 : 8;
    const int col = lane & 15;
    #pragma unroll
    for (int ct = 0; ct < 4; ++ct) {
        #pragma unroll
        for (int r = 0; r < 8; ++r) {
            op[(size_t)(r + row_off) * DIMW + ct * 16 + col] = accs[ct][r];
        }
    }
}

// ---------------- launcher ----------------

extern "C" void kernel_launch(void* const* d_in, const int* in_sizes, int n_in,
                              void* d_out, int out_size, void* d_ws, size_t ws_size,
                              hipStream_t stream) {
    const float* xpos = (const float*)d_in[0];   // input_positions  (2,4,4096,3)
    const float* wts  = (const float*)d_in[1];   // input_weights    (2,4,4096,64)
    const float* ypos = (const float*)d_in[2];   // output_positions (2,4,4096,3)
    float* out = (float*)d_out;                  // (2,4,4096,64)

    unsigned char* ws = (unsigned char*)d_ws;
    half_t* wT   = (half_t*)(ws);
    float*  xsoa = (float*)(ws + WS_XS_OFF);
    float4* ys4  = (float4*)(ws + WS_YS_OFF);

    prep_weights<<<(BC * DIMW * N_PTS) / 256, 256, 0, stream>>>(wts, wT);
    prep_pos_soa<<<(BC * N_PTS) / 256, 256, 0, stream>>>(xpos, xsoa);
    prep_pos_aos<<<(BC * N_PTS) / 256, 256, 0, stream>>>(ypos, ys4);

    dim3 grid(N_PTS / M_TILE, BC);               // 32 row-tiles x 8 (b,c)
    gauss_gram_wmma<<<grid, 256, 0, stream>>>(wT, xsoa, ys4, out);
}
